// GRU_87900800679944
// MI455X (gfx1250) — matrix operand order repaired and checked by reading, not compiled, not run on hardware
//
#include <hip/hip_runtime.h>
#include <hip/hip_bf16.h>
#include <math.h>

// ---------------- problem sizes ----------------
#define B_  512
#define T_  1024
#define D_  64
#define H_  256
#define A_  256
#define OD_ 16

#define BT        32            // batch rows per workgroup
#define NWG       (B_/BT)       // 16 workgroups
#define NTHREADS  256           // 8 wave32s
#define NWAVES    8
#define KC        (D_+H_)       // 320: combined K for [x_in | h]

// ---------------- LDS layout (bytes) ----------------
#define OFF_WCAT   0
#define SZ_WCAT    (H_*KC*2)            // 163840: [W_ih | W_hh] bf16, row n contiguous K=320
#define OFF_FC1    (OFF_WCAT+SZ_WCAT)
#define SZ_FC1     (A_*H_*2)            // 131072: fc1_w bf16 [256][256]
#define OFF_ABUF   (OFF_FC1+SZ_FC1)
#define SZ_ABUF    (BT*KC*2)            // 20480: A = [x_in(64) | h(256)] bf16 per batch row
#define OFF_SCR    (OFF_ABUF+SZ_ABUF)
#define SZ_SCR     (NWAVES*16*32*2)     // 8192: per-wave [16][32] bf16 fc2-input staging
#define OFF_OBUF   (OFF_SCR+SZ_SCR)
#define SZ_OBUF    (BT*OD_*4)           // 2048: f32 output accumulator
#define OFF_BC     (OFF_OBUF+SZ_OBUF)
#define SZ_BC      (H_*2)               // b_ih+b_hh, bf16
#define OFF_FB     (OFF_BC+SZ_BC)
#define SZ_FB      (A_*2)               // fc1_b, bf16
#define LDS_BYTES  (OFF_FB+SZ_FB)       // 326656 < 327680 (320 KB)

typedef __attribute__((ext_vector_type(16))) __bf16 v16bf;
typedef __attribute__((ext_vector_type(8)))  float  v8f;

__device__ __forceinline__ v8f wmma_bf16(v16bf a, v16bf b, v8f c) {
  // D = A(16x32 bf16) * B(32x16 bf16) + C(16x16 f32)
  return __builtin_amdgcn_wmma_f32_16x16x32_bf16(false, a, false, b, (short)0, c,
                                                 false, false);
}

// A-fragment (16x32 bf16) from row-major LDS tile (rows = M, contiguous K).
// Layout: lane m=lane&15, half=lane>>4; VGPR j: K=kb+2j+16*half
//  =>  u32 indices kb/2 + 8*half + j (j=0..7 contiguous -> 2x b128).
__device__ __forceinline__ v16bf load_a_frag(const __bf16* tileBase, int strideElems,
                                             int kb, int lane) {
  const int half = (lane >> 4) & 1;
  const int m    = lane & 15;
  const uint32_t* r32 = (const uint32_t*)(tileBase + m * strideElems);
  const int i0 = (kb >> 1) + 8 * half;
  union { v16bf v; uint4 q[2]; } u;
  u.q[0] = *(const uint4*)(r32 + i0);       // ds_load_b128
  u.q[1] = *(const uint4*)(r32 + i0 + 4);   // ds_load_b128
  return u.v;
}

// B-fragment (32x16 bf16) for C = A * W^T: B[k][n] = W[n][k], W row-major [N][K].
// Layout (contiguous K, same order as A): lane n=lane&15, half=lane>>4; VGPR j: K=kb+2j+16*half
//  => u32 indices kb/2 + 8*half + j (j=0..7 contiguous -> 2x b128).
__device__ __forceinline__ v16bf load_b_frag(const __bf16* w, int strideElems,
                                             int nBase, int kb, int lane) {
  const int half = (lane >> 4) & 1;
  const int n    = lane & 15;
  const uint32_t* r32 = (const uint32_t*)(w + (nBase + n) * strideElems);
  const int i0 = (kb >> 1) + 8 * half;
  union { v16bf v; uint4 q[2]; } u;
  u.q[0] = *(const uint4*)(r32 + i0);
  u.q[1] = *(const uint4*)(r32 + i0 + 4);
  return u.v;
}

__device__ __forceinline__ uint32_t pack2_bf16(float lo, float hi) {
  union { uint32_t u; __bf16 h[2]; } p;
  p.h[0] = (__bf16)lo; p.h[1] = (__bf16)hi;
  return p.u;
}

__global__ __launch_bounds__(NTHREADS)
void rnn_persistent(const float* __restrict__ x,
                    const float* __restrict__ W_ih, const float* __restrict__ W_hh,
                    const float* __restrict__ b_ih, const float* __restrict__ b_hh,
                    const float* __restrict__ fc1_w, const float* __restrict__ fc1_b,
                    const float* __restrict__ fc2_w, const float* __restrict__ fc2_b,
                    const int* __restrict__ psp,
                    float* __restrict__ out) {
  extern __shared__ char smem[];
  __bf16* Wc  = (__bf16*)(smem + OFF_WCAT);
  __bf16* F1  = (__bf16*)(smem + OFF_FC1);
  __bf16* Ab  = (__bf16*)(smem + OFF_ABUF);
  __bf16* Scr = (__bf16*)(smem + OFF_SCR);
  float*  Ob  = (float*)(smem + OFF_OBUF);
  __bf16* Bc  = (__bf16*)(smem + OFF_BC);
  __bf16* Fb  = (__bf16*)(smem + OFF_FB);

  const int tid  = threadIdx.x;
  const int lane = tid & 31;
  const int w    = tid >> 5;          // wave id 0..7
  const int half = (lane >> 4) & 1;
  const int nl   = lane & 15;
  const int row0 = blockIdx.x * BT;   // batch base row
  const int ps   = psp[0];

  // ---- one-time init: weights -> LDS as bf16 ----
  for (int i = tid; i < H_ * D_; i += NTHREADS) {
    int n = i >> 6, c = i & 63;
    Wc[n * KC + c] = (__bf16)W_ih[i];
  }
  for (int i = tid; i < H_ * H_; i += NTHREADS) {
    int n = i >> 8, c = i & 255;
    Wc[n * KC + D_ + c] = (__bf16)W_hh[i];
  }
  for (int i = tid; i < A_ * H_; i += NTHREADS) F1[i] = (__bf16)fc1_w[i];
  if (tid < H_) Bc[tid] = (__bf16)(b_ih[tid] + b_hh[tid]);
  if (tid < A_) Fb[tid] = (__bf16)fc1_b[tid];
  for (int i = tid; i < BT * H_; i += NTHREADS) {      // h0 = 0
    int m = i >> 8, c = i & 255;
    Ab[m * KC + D_ + c] = (__bf16)0.0f;
  }

  // fc2 B-fragment: per-wave K-slice [32w, 32w+32), held in 8 VGPRs for all T steps
  union { v16bf v; uint32_t u[8]; } f2;
  {
    const int kb = 32 * w;
#pragma unroll
    for (int j = 0; j < 8; ++j) {
      int k = kb + 2 * j + 16 * half;
      f2.u[j] = pack2_bf16(fc2_w[nl * H_ + k], fc2_w[nl * H_ + k + 1]);
    }
  }
  __syncthreads();

  const int   nt0  = 2 * w, nt1 = 2 * w + 1;   // this wave's N-tile pair
  const float bc0f = (float)Bc[nt0 * 16 + nl];
  const float bc1f = (float)Bc[nt1 * 16 + nl];
  const float fb0f = (float)Fb[nt0 * 16 + nl];
  const float fb1f = (float)Fb[nt1 * 16 + nl];
  __bf16* myScr = Scr + w * (16 * 32);

  for (int t = 0; t < T_; ++t) {
    // ---- stage 1: build x_in (bf16) into Ab cols [0,64) ----
    {
      const int m  = tid >> 3;
      const int cb = (tid & 7) * 8;
      const float* src = x + ((size_t)(row0 + m) * T_ + t) * D_ + cb;
      if (t > ps && cb < OD_) {            // autoregressive: feed previous output
#pragma unroll
        for (int c = 0; c < 8; ++c)
          Ab[m * KC + cb + c] = (__bf16)Ob[m * OD_ + cb + c];
      } else {
#pragma unroll
        for (int c = 0; c < 8; ++c)
          Ab[m * KC + cb + c] = (__bf16)src[c];
      }
      if (t + 1 < T_) __builtin_prefetch(src + D_, 0, 0);   // global_prefetch_b8
    }
    __syncthreads();

    // ---- stage 2: h_new = tanh([x|h] @ Wc^T + b),  M=32 N=256 K=320 ----
    v8f acc00 = {}, acc01 = {}, acc10 = {}, acc11 = {};
    for (int kf = 0; kf < KC / 32; ++kf) {
      const int kb = kf * 32;
      v16bf a0 = load_a_frag(Ab,            KC, kb, lane);   // rows 0..15
      v16bf a1 = load_a_frag(Ab + 16 * KC,  KC, kb, lane);   // rows 16..31
      v16bf b0 = load_b_frag(Wc, KC, nt0 * 16, kb, lane);
      v16bf b1 = load_b_frag(Wc, KC, nt1 * 16, kb, lane);
      acc00 = wmma_bf16(a0, b0, acc00);
      acc01 = wmma_bf16(a0, b1, acc01);
      acc10 = wmma_bf16(a1, b0, acc10);
      acc11 = wmma_bf16(a1, b1, acc11);
    }
    // re-init output accumulator with fc2 bias (Ob free: prev step consumed it)
    for (int e = tid; e < BT * OD_; e += NTHREADS) Ob[e] = fc2_b[e & (OD_ - 1)];
    // tanh + store h_new back into Ab cols [64,320)
    {
      const int n0 = nt0 * 16 + nl, n1 = nt1 * 16 + nl;
#pragma unroll
      for (int r = 0; r < 8; ++r) {
        const int m0 = r + 8 * half, m1 = 16 + r + 8 * half;
        Ab[m0 * KC + D_ + n0] = (__bf16)tanhf(acc00[r] + bc0f);
        Ab[m0 * KC + D_ + n1] = (__bf16)tanhf(acc01[r] + bc1f);
        Ab[m1 * KC + D_ + n0] = (__bf16)tanhf(acc10[r] + bc0f);
        Ab[m1 * KC + D_ + n1] = (__bf16)tanhf(acc11[r] + bc1f);
      }
    }
    __syncthreads();

    // ---- stage 3: fc1 pre-activation,  M=32 N=256 K=256 ----
    v8f f00 = {}, f01 = {}, f10 = {}, f11 = {};
    for (int kf = 0; kf < H_ / 32; ++kf) {
      const int kb = kf * 32;
      v16bf a0 = load_a_frag(Ab + D_,           KC, kb, lane);
      v16bf a1 = load_a_frag(Ab + 16 * KC + D_, KC, kb, lane);
      v16bf b0 = load_b_frag(F1, H_, nt0 * 16, kb, lane);
      v16bf b1 = load_b_frag(F1, H_, nt1 * 16, kb, lane);
      f00 = wmma_bf16(a0, b0, f00);
      f01 = wmma_bf16(a0, b1, f01);
      f10 = wmma_bf16(a1, b0, f10);
      f11 = wmma_bf16(a1, b1, f11);
    }

    // ---- stage 4: o = relu(fc1) @ fc2^T : stage this wave's K=32 slice,
    //      reload as A-frag, WMMA with resident fc2 frag, ds_add partials ----
    v8f oa0 = {}, oa1 = {};
#pragma unroll
    for (int mt = 0; mt < 2; ++mt) {
      v8f fa = mt ? f10 : f00;
      v8f fb = mt ? f11 : f01;
#pragma unroll
      for (int r = 0; r < 8; ++r) {
        const int mm = r + 8 * half;
        myScr[mm * 32 + nl]      = (__bf16)fmaxf(fa[r] + fb0f, 0.0f);
        myScr[mm * 32 + 16 + nl] = (__bf16)fmaxf(fb[r] + fb1f, 0.0f);
      }
      v16bf af = load_a_frag(myScr, 32, 0, lane);  // per-wave scratch, in-order DS
      if (mt == 0) oa0 = wmma_bf16(af, f2.v, oa0);
      else         oa1 = wmma_bf16(af, f2.v, oa1);
    }
#pragma unroll
    for (int r = 0; r < 8; ++r) {
      atomicAdd(&Ob[(r + 8 * half) * OD_ + nl],      oa0[r]);   // ds_add_f32
      atomicAdd(&Ob[(16 + r + 8 * half) * OD_ + nl], oa1[r]);
    }
    __syncthreads();

    // ---- stage 5: write o_t to global (f32, [B][T][OD]) ----
    for (int e = tid; e < BT * OD_; e += NTHREADS) {
      const int m = e >> 4, n = e & 15;
      out[((size_t)(row0 + m) * T_ + t) * OD_ + n] = Ob[e];
    }
    // no barrier needed: next stage-1 only reads Ob / writes Ab x-region
  }
}

extern "C" void kernel_launch(void* const* d_in, const int* in_sizes, int n_in,
                              void* d_out, int out_size, void* d_ws, size_t ws_size,
                              hipStream_t stream) {
  (void)in_sizes; (void)n_in; (void)out_size; (void)d_ws; (void)ws_size;
  const float* x     = (const float*)d_in[0];
  const float* W_ih  = (const float*)d_in[1];
  const float* W_hh  = (const float*)d_in[2];
  const float* b_ih  = (const float*)d_in[3];
  const float* b_hh  = (const float*)d_in[4];
  const float* fc1_w = (const float*)d_in[5];
  const float* fc1_b = (const float*)d_in[6];
  const float* fc2_w = (const float*)d_in[7];
  const float* fc2_b = (const float*)d_in[8];
  const int*   ps    = (const int*)d_in[9];
  float* out = (float*)d_out;

  hipFuncSetAttribute((const void*)rnn_persistent,
                      hipFuncAttributeMaxDynamicSharedMemorySize, LDS_BYTES);
  rnn_persistent<<<dim3(NWG), dim3(NTHREADS), LDS_BYTES, stream>>>(
      x, W_ih, W_hh, b_ih, b_hh, fc1_w, fc1_b, fc2_w, fc2_b, ps, out);
}
